// dummy_layer_20203526160416
// MI455X (gfx1250) — matrix operand order repaired and checked by reading, not compiled, not run on hardware
//
#include <hip/hip_runtime.h>

typedef __attribute__((ext_vector_type(16))) _Float16 v16h;
typedef __attribute__((ext_vector_type(8)))  float    v8f;
typedef __attribute__((ext_vector_type(4)))  float    v4f;

#define D 128          // D_IN == D_OUT
#define TWO_D 256

// ---------------------------------------------------------------------------
// Zero accumulator + degree buffer (ws is poisoned, must re-zero every call)
// ---------------------------------------------------------------------------
__global__ void k_zero(float* __restrict__ p, unsigned long long n) {
    unsigned long long i = (unsigned long long)blockIdx.x * blockDim.x + threadIdx.x;
    if (i < n) p[i] = 0.0f;
}

// ---------------------------------------------------------------------------
// Convert W [128 out, 256 in] f32  ->  Wh [256 k, 128 out] f16 (K-major).
// A B-fragment's 16 k values sit D halves (256B) apart; lanes read adjacent columns.
// ---------------------------------------------------------------------------
__global__ void k_convw(const float* __restrict__ W, _Float16* __restrict__ Wh) {
    int i = blockIdx.x * blockDim.x + threadIdx.x;   // 0 .. 256*128-1
    int k = i >> 7;          // 0..255
    int o = i & (D - 1);     // 0..127
    Wh[i] = (_Float16)W[o * TWO_D + k];              // Wh[k*128 + o] = W[o][k]
}

// ---------------------------------------------------------------------------
// Scatter: one wave32 per edge. lane handles features 4*lane..4*lane+3.
// agg[dst] += n_feats[src] * ew ; deg[dst] += 1 (lane 0).
// f32 atomics resolve in L2 (agg = 25.6MB, L2-resident on 192MB L2).
// ---------------------------------------------------------------------------
__global__ __launch_bounds__(256) void
k_scatter(const float* __restrict__ n_feats, const float* __restrict__ ew,
          const int* __restrict__ src, const int* __restrict__ dst,
          float* __restrict__ agg, float* __restrict__ deg, int n_edges) {
    unsigned long long gtid = (unsigned long long)blockIdx.x * blockDim.x + threadIdx.x;
    int edge = (int)(gtid >> 5);
    int lane = threadIdx.x & 31;
    if (edge >= n_edges) return;

    int   s = src[edge];
    int   d = dst[edge];
    float w = ew[edge];

    v4f v = *(const v4f*)(n_feats + (unsigned long long)s * D + lane * 4);
    float* dp = agg + (unsigned long long)d * D + lane * 4;
    atomicAdd(dp + 0, v.x * w);
    atomicAdd(dp + 1, v.y * w);
    atomicAdd(dp + 2, v.z * w);
    atomicAdd(dp + 3, v.w * w);
    if (lane == 0) atomicAdd(deg + d, 1.0f);
}

// ---------------------------------------------------------------------------
// GEMM: out[N,128] = concat(agg/max(deg,1), n_feats) @ Wh + b
// One wave per 16x128 output ROW-BLOCK: A fragment built once per K-chunk,
// reused by 8 WMMAs (one per 16-wide column tile). 64 wmma per wave.
//
// A frag (16-bit A 16x32): lane l -> M = l&15, kb = 8*(l>>4);
//   elems 0..7  = K = kb..kb+7,  elems 8..15 = K = kb+16..kb+23.
// B frag (16-bit B 32x16): lane l -> N = l&15, elems 0..15 = K in A's order (same kb).
// C/D (f32 16x16): vgpr r -> (M = r + 8*(l>>4), N = l&15).
// ---------------------------------------------------------------------------
__global__ __launch_bounds__(160) void
k_gemm(const float* __restrict__ agg, const float* __restrict__ deg,
       const float* __restrict__ n_feats, const _Float16* __restrict__ Wh,
       const float* __restrict__ bias, float* __restrict__ out, int n_row_tiles) {
    int wave = (int)((blockIdx.x * blockDim.x + threadIdx.x) >> 5); // = row tile
    int lane = threadIdx.x & 31;
    if (wave >= n_row_tiles) return;          // wave-uniform; EXEC stays all-1s

    int row0 = wave * 16;
    int m    = row0 + (lane & 15);
    int kb   = (lane >> 4) * 8;

    float rdeg = 1.0f / fmaxf(deg[m], 1.0f);
    const float* aggRow  = agg     + (unsigned long long)m * D;
    const float* featRow = n_feats + (unsigned long long)m * D;

    v8f acc[8] = {};
#pragma unroll
    for (int kc = 0; kc < 8; ++kc) {
        // K-chunk [kc*32, kc*32+32): kc<4 -> h = agg*rdeg, kc>=4 -> n_feats
        const float* sr = (kc < 4) ? aggRow : (featRow - D);   // index by global k
        float sc        = (kc < 4) ? rdeg : 1.0f;
        int k0 = kc * 32 + kb;

        v4f p0 = *(const v4f*)(sr + k0);
        v4f p1 = *(const v4f*)(sr + k0 + 4);
        v4f p2 = *(const v4f*)(sr + k0 + 16);
        v4f p3 = *(const v4f*)(sr + k0 + 20);

        v16h a;
#pragma unroll
        for (int j = 0; j < 4; ++j) {
            a[j]      = (_Float16)(p0[j] * sc);
            a[4 + j]  = (_Float16)(p1[j] * sc);
            a[8 + j]  = (_Float16)(p2[j] * sc);
            a[12 + j] = (_Float16)(p3[j] * sc);
        }

        // B column n = lane&15; elems follow A's K order (kb+s, kb+16+(s-8))
        const _Float16* brow = Wh + (unsigned long long)(kc * 32) * D + (lane & 15);
#pragma unroll
        for (int ct = 0; ct < 8; ++ct) {
            v16h bfrag;
#pragma unroll
            for (int s = 0; s < 16; ++s)
                bfrag[s] = brow[(kb + s + (s >= 8 ? 8 : 0)) * D + ct * 16];
            acc[ct] = __builtin_amdgcn_wmma_f32_16x16x32_f16(
                /*neg_a=*/false, a, /*neg_b=*/false, bfrag,
                /*c_mod=*/(short)0, acc[ct], /*reuse_a=*/false, /*reuse_b=*/false);
        }
    }

    int nloc  = lane & 15;
    int mbase = row0 + ((lane >> 4) << 3);
#pragma unroll
    for (int ct = 0; ct < 8; ++ct) {
        int   n    = ct * 16 + nloc;
        float bval = bias[n];
#pragma unroll
        for (int r = 0; r < 8; ++r)
            out[(unsigned long long)(mbase + r) * D + n] = acc[ct][r] + bval;
    }
}

// ---------------------------------------------------------------------------
extern "C" void kernel_launch(void* const* d_in, const int* in_sizes, int n_in,
                              void* d_out, int out_size, void* d_ws, size_t ws_size,
                              hipStream_t stream) {
    const float* n_feats = (const float*)d_in[0];
    const float* ew      = (const float*)d_in[1];
    const float* W       = (const float*)d_in[2];
    const float* b       = (const float*)d_in[3];
    const int*   src     = (const int*)d_in[4];
    const int*   dst     = (const int*)d_in[5];
    float*       out     = (float*)d_out;

    int n_nodes = in_sizes[0] / D;     // 50000
    int n_edges = in_sizes[4];         // 600000

    // Workspace layout: agg [N*128] f32 | deg [N] f32 | Wh [256*128] f16 (256B aligned)
    char*  ws    = (char*)d_ws;
    float* agg   = (float*)ws;
    float* deg   = agg + (unsigned long long)n_nodes * D;
    unsigned long long zcount = (unsigned long long)n_nodes * D + n_nodes;
    unsigned long long wh_off = (zcount * sizeof(float) + 255ull) & ~255ull;
    _Float16* Wh = (_Float16*)(ws + wh_off);

    // 1) zero accumulators
    unsigned zb = (unsigned)((zcount + 1023) / 1024);
    k_zero<<<zb, 1024, 0, stream>>>(agg, zcount);

    // 2) convert + transpose W -> f16 K-major
    k_convw<<<(TWO_D * D) / 256, 256, 0, stream>>>(W, Wh);

    // 3) scatter-add messages + degree (one wave per edge)
    unsigned long long sthreads = (unsigned long long)n_edges * 32ull;
    k_scatter<<<(unsigned)((sthreads + 255) / 256), 256, 0, stream>>>(
        n_feats, ew, src, dst, agg, deg, n_edges);

    // 4) WMMA GEMM: one wave per 16-row block; 3125 = 5^5 row tiles,
    //    160 threads = 5 waves/block -> 625 blocks, zero tail.
    int n_row_tiles = n_nodes / 16;                 // 3125
    unsigned gblocks = (unsigned)((n_row_tiles + 4) / 5);   // 625
    k_gemm<<<gblocks, 160, 0, stream>>>(agg, deg, n_feats, Wh, b, out, n_row_tiles);
}
